// SpatialCGNLx_79740362817676
// MI455X (gfx1250) — compile-verified, matrix operand order checked by reading, not run on hardware
//
#include <hip/hip_runtime.h>
#include <hip/hip_bf16.h>

typedef __attribute__((ext_vector_type(16))) __bf16       v16bf;
typedef __attribute__((ext_vector_type(8)))  float        v8f;
typedef __attribute__((ext_vector_type(8)))  unsigned int v8u;

union FragU { v8u u; v16bf b; };

#define BATCH 16
#define CIN   1024
#define PL    512
#define HW    1024
#define NG    8

// workspace byte offsets
#define WT_OFF   ((size_t)0)
#define WP_OFF   ((size_t)1 << 20)
#define WG_OFF   ((size_t)2 << 20)
#define WZ_OFF   ((size_t)3 << 20)
#define T_OFF    ((size_t)4 << 20)          // 32 MB fp32
#define Z_OFF    ((size_t)36 << 20)         // 64 MB fp32
#define ATT_OFF  ((size_t)100 << 20)        // 512 floats
#define COEF_OFF (ATT_OFF + 4096)           // 512 floats
#define GN_OFF   (ATT_OFF + 8192)           // 256 floats

__device__ __forceinline__ unsigned short f2bf(float f) {
  unsigned int u = __float_as_uint(f);
  u += 0x7FFFu + ((u >> 16) & 1u);          // round-to-nearest-even
  return (unsigned short)(u >> 16);
}
__device__ __forceinline__ unsigned int pack2bf(float a, float b) {
  return (unsigned int)f2bf(a) | ((unsigned int)f2bf(b) << 16);
}
// dword offset of bf16 K-pair e2 (0..7) within a 32-wide K step, per lane half
__device__ __forceinline__ int kpair(int e2, int half) {
  return ((e2 >> 2) << 3) + (half << 2) + (e2 & 3);
}

__global__ void k_init(float* att, float* gn) {
  int i = threadIdx.x;
  if (i < 512) att[i] = 0.f;
  if (i < 256) gn[i] = 0.f;
}

__global__ void k_cvt_bf16(const float* __restrict__ src,
                           unsigned short* __restrict__ dst, int n) {
  int i = blockIdx.x * blockDim.x + threadIdx.x;
  if (i < n) dst[i] = f2bf(src[i]);
}

// ---- fused t/p/g GEMM: 128x64 tile of all three outputs per workgroup ----
// Double-buffered LDS staging: next B tile's global loads issue before the
// current tile's WMMAs; one barrier per K-step.
__global__ __launch_bounds__(256) void k_gemm_tpg(
    const float* __restrict__ x, const char* __restrict__ wsb,
    float* __restrict__ t_ws, float* __restrict__ att)
{
  __shared__ unsigned int bTile[2][64 * 16]; // [N=64][K=32] bf16, K contiguous
  const unsigned int* Wu0 = (const unsigned int*)(wsb + WT_OFF);
  const unsigned int* Wu1 = (const unsigned int*)(wsb + WP_OFF);
  const unsigned int* Wu2 = (const unsigned int*)(wsb + WG_OFF);
  const unsigned int* Wu[3] = {Wu0, Wu1, Wu2};

  const int b     = blockIdx.z;
  const int mbase = blockIdx.y * 128;
  const int nbase = blockIdx.x * 64;
  const int tid   = threadIdx.x;
  const int lane  = tid & 31, wave = tid >> 5;
  const int half  = lane >> 4, l16 = lane & 15;

  v8f acc[3][4] = {};

  const float* xb = x + (size_t)b * CIN * HW;
  const int sn = tid & 63;                  // staging N column
  const int kq = tid >> 6;                  // 0..3 (K sub-chunk of 8)

  auto stage = [&](int kk, int buf) {
    const float* src = xb + (size_t)(kk * 32 + kq * 8) * HW + nbase + sn;
#pragma unroll
    for (int i = 0; i < 4; ++i) {
      float a = src[(size_t)(2 * i) * HW];
      float c = src[(size_t)(2 * i + 1) * HW];
      bTile[buf][sn * 16 + kq * 4 + i] = pack2bf(a, c);
    }
  };

  stage(0, 0);
  __syncthreads();

  for (int kk = 0; kk < CIN / 32; ++kk) {
    const int buf = kk & 1;
    if (kk + 1 < CIN / 32) stage(kk + 1, buf ^ 1);   // prefetch next tile

    FragU bfv[4];
#pragma unroll
    for (int nt = 0; nt < 4; ++nt)
#pragma unroll
      for (int e2 = 0; e2 < 8; ++e2)
        bfv[nt].u[e2] = bTile[buf][(nt * 16 + l16) * 16 + kpair(e2, half)];

#pragma unroll
    for (int m = 0; m < 3; ++m) {
      FragU af;
      const int row = mbase + wave * 16 + l16;
#pragma unroll
      for (int e2 = 0; e2 < 8; ++e2)
        af.u[e2] = Wu[m][row * 512 + kk * 16 + kpair(e2, half)];
#pragma unroll
      for (int nt = 0; nt < 4; ++nt)
        acc[m][nt] = __builtin_amdgcn_wmma_f32_16x16x32_bf16(
            false, af.b, false, bfv[nt].b, (short)0, acc[m][nt], false, false);
    }
    __syncthreads();
  }

  // store t (fp32)
#pragma unroll
  for (int nt = 0; nt < 4; ++nt)
#pragma unroll
    for (int v = 0; v < 8; ++v) {
      int row = mbase + wave * 16 + v + half * 8;
      int col = nbase + nt * 16 + l16;
      t_ws[(size_t)b * PL * HW + (size_t)row * HW + col] = acc[0][nt][v];
    }

  // attention raw sums: sum g, p*g, p^2*g, p^3*g  (p,g never hit HBM)
  float s0 = 0.f, s1 = 0.f, s2 = 0.f, s3 = 0.f;
#pragma unroll
  for (int nt = 0; nt < 4; ++nt)
#pragma unroll
    for (int v = 0; v < 8; ++v) {
      float pv = acc[1][nt][v], gv = acc[2][nt][v];
      s0 += gv;
      float pp = pv;  s1 += pp * gv;
      pp *= pv;       s2 += pp * gv;
      pp *= pv;       s3 += pp * gv;
    }
#pragma unroll
  for (int m = 16; m >= 1; m >>= 1) {
    s0 += __shfl_xor(s0, m, 32);
    s1 += __shfl_xor(s1, m, 32);
    s2 += __shfl_xor(s2, m, 32);
    s3 += __shfl_xor(s3, m, 32);
  }
  if (lane == 0) {
    int grp = (mbase + wave * 16) >> 6;     // 64 t-channels per group
    float* a = att + ((size_t)b * NG + grp) * 4;
    atomicAdd(a + 0, s0); atomicAdd(a + 1, s1);
    atomicAdd(a + 2, s2); atomicAdd(a + 3, s3);
  }
}

// coef[b,g,o] = alpha_o^2 * raw_o   (alpha_o^2 = (2g)^o/o! * exp(-2g))
__global__ void k_coef(const float* __restrict__ att, float* __restrict__ coef) {
  const float A2[4] = {0.99980001999866673f, 1.9996000399973335e-4f,
                       1.9996000399973335e-8f, 1.3330666933315556e-12f};
  int i = threadIdx.x;                      // 512
  coef[i] = att[i] * A2[i & 3];
}

// ---- y = poly(t) -> grouped conv z (WMMA) -> z + groupnorm sums ----
__global__ __launch_bounds__(256) void k_zgemm(
    const char* __restrict__ wsb, const float* __restrict__ t_ws,
    const float* __restrict__ coef, float* __restrict__ z_ws,
    float* __restrict__ gn)
{
  __shared__ unsigned int yTile[128 * 32];  // [N=128][K=64] bf16, K contiguous
  const unsigned int* Wzu = (const unsigned int*)(wsb + WZ_OFF);
  const int b = blockIdx.z, g = blockIdx.y;
  const int hwbase = blockIdx.x * 128;
  const int tid  = threadIdx.x, lane = tid & 31, wave = tid >> 5;
  const int half = lane >> 4, l16 = lane & 15;

  const float* cf = coef + ((size_t)b * NG + g) * 4;
  const float c0 = cf[0], c1 = cf[1], c2 = cf[2], c3 = cf[3];

  // stage y tile: evaluate Taylor poly of t on the fly, bf16 into LDS
  {
    int n = tid & 127, kh = tid >> 7;       // kh in {0,1}
    const float* tsrc = t_ws + (size_t)b * PL * HW +
                        (size_t)(g * 64 + kh * 32) * HW + hwbase + n;
#pragma unroll
    for (int i = 0; i < 16; ++i) {
      float ta = tsrc[(size_t)(2 * i) * HW];
      float tb = tsrc[(size_t)(2 * i + 1) * HW];
      float ya = c0 + ta * (c1 + ta * (c2 + ta * c3));
      float yb = c0 + tb * (c1 + tb * (c2 + tb * c3));
      yTile[n * 32 + kh * 16 + i] = pack2bf(ya, yb);
    }
  }
  __syncthreads();

  v8f acc[8] = {};
#pragma unroll
  for (int ks = 0; ks < 2; ++ks) {
    FragU af;
    const int row = wave * 16 + l16;        // oc row
#pragma unroll
    for (int e2 = 0; e2 < 8; ++e2)
      af.u[e2] = Wzu[(size_t)(g * 128 + row) * 32 + ks * 16 + kpair(e2, half)];
#pragma unroll
    for (int nt = 0; nt < 8; ++nt) {
      FragU bfv;
#pragma unroll
      for (int e2 = 0; e2 < 8; ++e2)
        bfv.u[e2] = yTile[(nt * 16 + l16) * 32 + ks * 16 + kpair(e2, half)];
      acc[nt] = __builtin_amdgcn_wmma_f32_16x16x32_bf16(
          false, af.b, false, bfv.b, (short)0, acc[nt], false, false);
    }
  }

  float s1 = 0.f, s2 = 0.f;
#pragma unroll
  for (int nt = 0; nt < 8; ++nt)
#pragma unroll
    for (int v = 0; v < 8; ++v) {
      int row = wave * 16 + v + half * 8;
      int col = hwbase + nt * 16 + l16;
      float zv = acc[nt][v];
      z_ws[(size_t)b * CIN * HW + (size_t)(g * 128 + row) * HW + col] = zv;
      s1 += zv; s2 += zv * zv;
    }
#pragma unroll
  for (int m = 16; m >= 1; m >>= 1) {
    s1 += __shfl_xor(s1, m, 32);
    s2 += __shfl_xor(s2, m, 32);
  }
  if (lane == 0) {
    float* p = gn + ((size_t)b * NG + g) * 2;
    atomicAdd(p + 0, s1); atomicAdd(p + 1, s2);
  }
}

// ---- groupnorm apply + affine + residual (float4 vectorized) ----
__global__ void k_final(const float4* __restrict__ z4, const float4* __restrict__ x4,
                        const float* __restrict__ gn, const float* __restrict__ gw,
                        const float* __restrict__ gb, float4* __restrict__ out4)
{
  int idx = blockIdx.x * 256 + threadIdx.x; // 4M threads, 4 hw elems each
  int e = idx << 2;                          // flat element index
  int b = e >> 20;
  int c = (e >> 10) & 1023;                  // 4 consecutive hw share channel c
  int g = c >> 7;
  const float* s = gn + ((size_t)b * NG + g) * 2;
  const float invN = 1.0f / 131072.0f;       // 128 ch * 1024 hw per group
  float mean = s[0] * invN;
  float var  = s[1] * invN - mean * mean;
  float inv  = rsqrtf(var + 1e-5f);
  float w = inv * gw[c];
  float bias = gb[c] - mean * w;
  float4 z = z4[idx];
  float4 xv = x4[idx];
  float4 o;
  o.x = z.x * w + bias + xv.x;
  o.y = z.y * w + bias + xv.y;
  o.z = z.z * w + bias + xv.z;
  o.w = z.w * w + bias + xv.w;
  out4[idx] = o;
}

extern "C" void kernel_launch(void* const* d_in, const int* in_sizes, int n_in,
                              void* d_out, int out_size, void* d_ws, size_t ws_size,
                              hipStream_t stream) {
  const float* x  = (const float*)d_in[0];
  const float* Wt = (const float*)d_in[1];
  const float* Wp = (const float*)d_in[2];
  const float* Wg = (const float*)d_in[3];
  const float* Wz = (const float*)d_in[4];
  const float* gw = (const float*)d_in[5];
  const float* gb = (const float*)d_in[6];
  char*  ws   = (char*)d_ws;
  float* att  = (float*)(ws + ATT_OFF);
  float* coef = (float*)(ws + COEF_OFF);
  float* gn   = (float*)(ws + GN_OFF);
  float* t_ws = (float*)(ws + T_OFF);
  float* z_ws = (float*)(ws + Z_OFF);

  k_init<<<1, 512, 0, stream>>>(att, gn);
  k_cvt_bf16<<<(PL * CIN + 255) / 256, 256, 0, stream>>>(Wt, (unsigned short*)(ws + WT_OFF), PL * CIN);
  k_cvt_bf16<<<(PL * CIN + 255) / 256, 256, 0, stream>>>(Wp, (unsigned short*)(ws + WP_OFF), PL * CIN);
  k_cvt_bf16<<<(PL * CIN + 255) / 256, 256, 0, stream>>>(Wg, (unsigned short*)(ws + WG_OFF), PL * CIN);
  k_cvt_bf16<<<(65536 + 255) / 256, 256, 0, stream>>>(Wz, (unsigned short*)(ws + WZ_OFF), 65536);

  k_gemm_tpg<<<dim3(16, 4, 16), 256, 0, stream>>>(x, ws, t_ws, att);
  k_coef<<<1, 512, 0, stream>>>(att, coef);
  k_zgemm<<<dim3(8, 8, 16), 256, 0, stream>>>(ws, t_ws, coef, z_ws, gn);
  k_final<<<(BATCH * CIN * HW / 4 + 255) / 256, 256, 0, stream>>>(
      (const float4*)z_ws, (const float4*)x, gn, gw, gb, (float4*)d_out);
}
